// IntermediateLayers_15745350107915
// MI455X (gfx1250) — compile-verified, matrix operand order checked by reading, not run on hardware
//
#include <hip/hip_runtime.h>
#include <hip/hip_bf16.h>
#include <math.h>

// ---------------- problem constants (from reference) ----------------
#define NEL   20      // electrons per sample
#define NUPE  10      // spin-up electrons
#define NATOM 4
#define HE    256
#define HEE   32
#define FIN   832     // 3*HE + 2*HEE
#define NLAY  4

typedef __bf16 bf16_t;
typedef __attribute__((ext_vector_type(16))) __bf16 v16bf;
typedef __attribute__((ext_vector_type(8)))  float  v8f;

// ---------------- gfx1250 feature detection ----------------
#if defined(__AMDGCN__) && __has_builtin(__builtin_amdgcn_global_load_async_to_lds_b128) && \
    __has_builtin(__builtin_amdgcn_s_wait_asynccnt)
#define HAVE_ASYNC 1
#else
#define HAVE_ASYNC 0
#endif

#if HAVE_ASYNC
// Builtin expects: (v4i addrspace(1)*, v4i addrspace(3)*, imm offset, imm cpol)
typedef int b128i __attribute__((vector_size(16)));
typedef __attribute__((address_space(1))) b128i gas_b128;
typedef __attribute__((address_space(3))) b128i las_b128;
__device__ __forceinline__ void async_copy16(const void* g, void* l) {
  // GLOBAL_LOAD_ASYNC_TO_LDS_B128: 16 bytes/lane, tracked by ASYNCcnt
  __builtin_amdgcn_global_load_async_to_lds_b128((gas_b128*)g, (las_b128*)l, 0, 0);
}
#endif

__device__ __forceinline__ float act_tanh(float x) {
#if defined(__AMDGCN__) && __has_builtin(__builtin_amdgcn_tanhf)
  return __builtin_amdgcn_tanhf(x);   // native V_TANH_F32 (gfx1250 TRANS op)
#else
  return tanhf(x);
#endif
}

// ---------------- WMMA helpers (CDNA5 gfx1250, wave32) ----------------
__device__ __forceinline__ v8f wmma_bf16(v16bf a, v16bf b, v8f c) {
  // 8 args: (neg_a, A, neg_b, B, c_mod, C, reuse_a, reuse_b)
  return __builtin_amdgcn_wmma_f32_16x16x32_bf16(false, a, false, b, (short)0, c,
                                                 false, false);
}

// A fragment: 16x32 bf16 from LDS (row-major, leading dim ld), ISA 16-bit layout.
__device__ __forceinline__ v16bf load_a_lds(const bf16_t* sA, int ld, int lane) {
  const int m = lane & 15, h = lane >> 4;
  v16bf a;
#pragma unroll
  for (int i = 0; i < 8; ++i) {
    const int k = ((i & 3) << 1) + ((i >> 2) << 4) + (h << 3);
    a[2 * i]     = sA[m * ld + k];
    a[2 * i + 1] = sA[m * ld + k + 1];
  }
  return a;
}

// B fragment: weights pre-packed so each lane's 16 bf16 values are contiguous.
__device__ __forceinline__ v16bf load_b_packed(const bf16_t* __restrict__ Wp,
                                               int chunk, int ntile, int ntiles,
                                               int lane) {
  return *(const v16bf*)(Wp + (size_t)(((chunk * ntiles) + ntile) * 32 + lane) * 16);
}

// ---------------- weight packing: fp32 [Ksrc, Nout] -> bf16 fragment order ----
__global__ __launch_bounds__(256) void pack_w_kernel(const float* __restrict__ src,
                                                     bf16_t* __restrict__ dst,
                                                     int Ksrc, int Kpad, int Nout) {
  int idx = blockIdx.x * 256 + threadIdx.x;
  int total = Kpad * Nout;
  if (idx >= total) return;
  int e  = idx & 15;
  int l  = (idx >> 4) & 31;
  int ct = idx >> 9;
  int ntiles = Nout >> 4;
  int t = ct % ntiles;
  int chunk = ct / ntiles;
  int i = e >> 1;
  int k = chunk * 32 + ((i & 3) << 1) + ((i >> 2) << 4) + ((l >> 4) << 3) + (e & 1);
  int n = t * 16 + (l & 15);
  float v = (k < Ksrc) ? src[(size_t)k * Nout + n] : 0.0f;
  dst[idx] = (bf16_t)v;
}

// ---------------- layer-0 mean pools (from raw positions) ----------------
__global__ __launch_bounds__(256) void means0_hi_kernel(const float* __restrict__ pos,
                                                        const float* __restrict__ rnuc,
                                                        float* __restrict__ gdown,
                                                        float* __restrict__ gup,
                                                        int B) {
  int idx = blockIdx.x * 256 + threadIdx.x;
  if (idx >= B * 16) return;
  int b = idx >> 4, c = idx & 15;
  int a = c >> 2, comp = c & 3;
  float nx = rnuc[a * 3 + 0], ny = rnuc[a * 3 + 1], nz = rnuc[a * 3 + 2];
  float sd = 0.f, su = 0.f;
  for (int i = 0; i < NEL; ++i) {
    const float* p = pos + (size_t)(b * NEL + i) * 3;
    float dx = p[0] - nx, dy = p[1] - ny, dz = p[2] - nz;
    float f = (comp == 0) ? dx : (comp == 1) ? dy : (comp == 2) ? dz
              : sqrtf(dx * dx + dy * dy + dz * dz);
    if (i < NUPE) sd += f; else su += f;
  }
  gdown[idx] = sd * (1.0f / NUPE);
  gup[idx]   = su * (1.0f / (NEL - NUPE));
}

__global__ __launch_bounds__(256) void means0_hij_kernel(const float* __restrict__ pos,
                                                         float* __restrict__ gdni,
                                                         float* __restrict__ gupi,
                                                         int B) {
  int idx = blockIdx.x * 256 + threadIdx.x;
  if (idx >= B * NEL * 4) return;
  int b = idx / (NEL * 4);
  int rem = idx % (NEL * 4);
  int j = rem >> 2, c = rem & 3;
  const float* pj = pos + (size_t)(b * NEL + j) * 3;
  float jx = pj[0], jy = pj[1], jz = pj[2];
  float sd = 0.f, su = 0.f;
  for (int i = 0; i < NEL; ++i) {
    const float* pi = pos + (size_t)(b * NEL + i) * 3;
    float dx = pi[0] - jx, dy = pi[1] - jy, dz = pi[2] - jz;
    float f;
    if (c == 0) f = dx;
    else if (c == 1) f = dy;
    else if (c == 2) f = dz;
    else { float sq = dx * dx + dy * dy + dz * dz; f = (sq > 0.f) ? sqrtf(sq) : 0.f; }
    if (i < NUPE) sd += f; else su += f;
  }
  gdni[idx] = sd * (1.0f / NUPE);
  gupi[idx] = su * (1.0f / (NEL - NUPE));
}

// ---------------- later-layer mean pools ----------------
__global__ __launch_bounds__(256) void means_hi_kernel(const float* __restrict__ hi,
                                                       float* __restrict__ gdown,
                                                       float* __restrict__ gup,
                                                       int B) {
  int idx = blockIdx.x * 256 + threadIdx.x;
  if (idx >= B * HE) return;
  int b = idx >> 8, c = idx & 255;
  float sd = 0.f, su = 0.f;
  for (int i = 0; i < NEL; ++i) {
    float v = hi[(size_t)(b * NEL + i) * HE + c];
    if (i < NUPE) sd += v; else su += v;
  }
  gdown[idx] = sd * (1.0f / NUPE);
  gup[idx]   = su * (1.0f / (NEL - NUPE));
}

__global__ __launch_bounds__(256) void means_hij_kernel(const float* __restrict__ hij,
                                                        float* __restrict__ gdni,
                                                        float* __restrict__ gupi,
                                                        int B) {
  int idx = blockIdx.x * 256 + threadIdx.x;
  if (idx >= B * NEL * HEE) return;
  int c = idx & 31;
  int j = (idx >> 5) % NEL;
  int b = idx / (NEL * HEE);
  float sd = 0.f, su = 0.f;
  for (int i = 0; i < NEL; ++i) {
    float v = hij[(size_t)((b * NEL + i) * NEL + j) * HEE + c];
    if (i < NUPE) sd += v; else su += v;
  }
  gdni[idx] = sd * (1.0f / NUPE);
  gupi[idx] = su * (1.0f / (NEL - NUPE));
}

// ---------------- layer-0 one-electron GEMM: hi = tanh(f0 @ W + b) ----------
__global__ __launch_bounds__(256) void gemm_hi_l0_kernel(
    const float* __restrict__ pos, const float* __restrict__ rnuc,
    const float* __restrict__ gdown, const float* __restrict__ gup,
    const float* __restrict__ gdni, const float* __restrict__ gupi,
    const bf16_t* __restrict__ Wp, const float* __restrict__ bias,
    float* __restrict__ hi) {
  const int LD = 72;
  __shared__ bf16_t sA[16 * 72];
  const int tid = threadIdx.x;
  const int r0 = blockIdx.x * 16;
  for (int idx = tid; idx < 16 * 64; idx += 256) {
    int r = idx >> 6, c = idx & 63;
    int grow = r0 + r, b = grow / NEL, n = grow % NEL;
    float v = 0.f;
    if (c < 16) {
      int a = c >> 2, comp = c & 3;
      const float* p = pos + (size_t)grow * 3;
      float dx = p[0] - rnuc[a * 3 + 0], dy = p[1] - rnuc[a * 3 + 1],
            dz = p[2] - rnuc[a * 3 + 2];
      v = (comp == 0) ? dx : (comp == 1) ? dy : (comp == 2) ? dz
          : sqrtf(dx * dx + dy * dy + dz * dz);
    } else if (c < 32) v = gdown[b * 16 + (c - 16)];
    else if (c < 48)  v = gup[b * 16 + (c - 32)];
    else if (c < 52)  v = gdni[(b * NEL + n) * 4 + (c - 48)];
    else if (c < 56)  v = gupi[(b * NEL + n) * 4 + (c - 52)];
    sA[r * LD + c] = (bf16_t)v;
  }
  __syncthreads();
  const int wave = tid >> 5, lane = tid & 31;
  const int n0 = wave * 32;
  v8f acc0 = {0.f, 0.f, 0.f, 0.f, 0.f, 0.f, 0.f, 0.f};
  v8f acc1 = acc0;
#pragma unroll
  for (int c = 0; c < 2; ++c) {
    v16bf a  = load_a_lds(sA + c * 32, LD, lane);
    v16bf b0 = load_b_packed(Wp, c, 2 * wave,     16, lane);
    v16bf b1 = load_b_packed(Wp, c, 2 * wave + 1, 16, lane);
    acc0 = wmma_bf16(a, b0, acc0);
    acc1 = wmma_bf16(a, b1, acc1);
  }
  const int moff = (lane >> 4) << 3, nn = lane & 15;
  float bi0 = bias[n0 + nn], bi1 = bias[n0 + 16 + nn];
#pragma unroll
  for (int r = 0; r < 8; ++r) {
    size_t o = (size_t)(r0 + r + moff) * HE + n0 + nn;
    hi[o]      = act_tanh(acc0[r] + bi0);
    hi[o + 16] = act_tanh(acc1[r] + bi1);
  }
}

// ---------------- layers 1..3 one-electron GEMM (K=832, residual) ------------
// f-tile = [hi row | gdown[b] | gup[b] | gdni[b,n] | gupi[b,n]] gathered as
// row-contiguous fp32 segments -> async-copied straight into LDS, then one
// LDS->LDS bf16 conversion pass; residual comes from the fp32 staging tile.
__global__ __launch_bounds__(256) void gemm_hi_kernel(
    float* __restrict__ hi, const float* __restrict__ gdown,
    const float* __restrict__ gup, const float* __restrict__ gdni,
    const float* __restrict__ gupi, const bf16_t* __restrict__ Wp,
    const float* __restrict__ bias) {
  const int LDB = 848;                 // bf16 tile leading dim (FIN + 16 pad)
  __shared__ float  sAf[16 * FIN];     // fp32 staging (also residual source)
  __shared__ bf16_t sAb[16 * 848];     // bf16 WMMA tile
  const int tid = threadIdx.x;
  const int r0 = blockIdx.x * 16;

#if HAVE_ASYNC
  // 16 rows x 208 16-byte segments; LDS column == 4*cc by construction.
  for (int t = tid; t < 16 * 208; t += 256) {
    int r = t / 208, cc = t % 208;
    int grow = r0 + r, b = grow / NEL, n = grow % NEL;
    const float* g;
    if (cc < 64)       g = hi    + (size_t)grow * HE + cc * 4;
    else if (cc < 128) g = gdown + (size_t)b * HE + (cc - 64) * 4;
    else if (cc < 192) g = gup   + (size_t)b * HE + (cc - 128) * 4;
    else if (cc < 200) g = gdni  + (size_t)(b * NEL + n) * HEE + (cc - 192) * 4;
    else               g = gupi  + (size_t)(b * NEL + n) * HEE + (cc - 200) * 4;
    async_copy16(g, sAf + r * FIN + cc * 4);
  }
  __builtin_amdgcn_s_wait_asynccnt(0);
#else
  for (int idx = tid; idx < 16 * FIN; idx += 256) {
    int r = idx / FIN, c = idx % FIN;
    int grow = r0 + r, b = grow / NEL, n = grow % NEL;
    float v;
    if (c < 256)      v = hi[(size_t)grow * HE + c];
    else if (c < 512) v = gdown[b * HE + (c - 256)];
    else if (c < 768) v = gup[b * HE + (c - 512)];
    else if (c < 800) v = gdni[(b * NEL + n) * HEE + (c - 768)];
    else              v = gupi[(b * NEL + n) * HEE + (c - 800)];
    sAf[idx] = v;
  }
#endif
  __syncthreads();
  for (int idx = tid; idx < 16 * FIN; idx += 256) {
    int r = idx / FIN, c = idx % FIN;
    sAb[r * LDB + c] = (bf16_t)sAf[idx];
  }
  __syncthreads();

  const int wave = tid >> 5, lane = tid & 31;
  const int n0 = wave * 32;
  v8f acc0 = {0.f, 0.f, 0.f, 0.f, 0.f, 0.f, 0.f, 0.f};
  v8f acc1 = acc0;
  for (int c = 0; c < FIN / 32; ++c) {   // 26 K-chunks
    v16bf a  = load_a_lds(sAb + c * 32, LDB, lane);
    v16bf b0 = load_b_packed(Wp, c, 2 * wave,     16, lane);
    v16bf b1 = load_b_packed(Wp, c, 2 * wave + 1, 16, lane);
    // prefetch next chunk's packed weights (global_prefetch_b8)
    int cn = (c + 1 < FIN / 32) ? (c + 1) : c;
    __builtin_prefetch(Wp + (size_t)((cn * 16 + 2 * wave) * 32 + lane) * 16, 0, 3);
    acc0 = wmma_bf16(a, b0, acc0);
    acc1 = wmma_bf16(a, b1, acc1);
  }
  const int moff = (lane >> 4) << 3, nn = lane & 15;
  float bi0 = bias[n0 + nn], bi1 = bias[n0 + 16 + nn];
#pragma unroll
  for (int r = 0; r < 8; ++r) {
    size_t o = (size_t)(r0 + r + moff) * HE + n0 + nn;
    float old0 = sAf[(r + moff) * FIN + n0 + nn];
    float old1 = sAf[(r + moff) * FIN + n0 + 16 + nn];
    hi[o]      = act_tanh(acc0[r] + bi0) + old0;
    hi[o + 16] = act_tanh(acc1[r] + bi1) + old1;
  }
}

// ---------------- layer-0 two-electron GEMM (features inline, K pad 4->32) ---
__global__ __launch_bounds__(256) void gemm_hij_l0_kernel(
    const float* __restrict__ pos, const bf16_t* __restrict__ Wp,
    const float* __restrict__ bias, float* __restrict__ hij) {
  const int LD = 40;
  __shared__ bf16_t sA[128 * 40];
  const int tid = threadIdx.x;
  const int r0 = blockIdx.x * 128;
  for (int idx = tid; idx < 128 * 32; idx += 256) {
    int r = idx >> 5, c = idx & 31;
    float v = 0.f;
    if (c < 4) {
      int grow = r0 + r;
      int b = grow / (NEL * NEL);
      int rem = grow % (NEL * NEL);
      int i = rem / NEL, j = rem % NEL;
      const float* pi = pos + (size_t)(b * NEL + i) * 3;
      const float* pj = pos + (size_t)(b * NEL + j) * 3;
      float dx = pi[0] - pj[0], dy = pi[1] - pj[1], dz = pi[2] - pj[2];
      if (c == 0) v = dx;
      else if (c == 1) v = dy;
      else if (c == 2) v = dz;
      else { float sq = dx * dx + dy * dy + dz * dz; v = (sq > 0.f) ? sqrtf(sq) : 0.f; }
    }
    sA[r * LD + c] = (bf16_t)v;
  }
  __syncthreads();
  const int wave = tid >> 5, lane = tid & 31;
  v16bf a  = load_a_lds(sA + (wave * 16) * LD, LD, lane);
  v16bf b0 = load_b_packed(Wp, 0, 0, 2, lane);
  v16bf b1 = load_b_packed(Wp, 0, 1, 2, lane);
  v8f acc0 = {0.f, 0.f, 0.f, 0.f, 0.f, 0.f, 0.f, 0.f};
  v8f acc1 = acc0;
  acc0 = wmma_bf16(a, b0, acc0);
  acc1 = wmma_bf16(a, b1, acc1);
  const int moff = (lane >> 4) << 3, nn = lane & 15;
  float bi0 = bias[nn], bi1 = bias[16 + nn];
#pragma unroll
  for (int r = 0; r < 8; ++r) {
    size_t o = (size_t)(r0 + wave * 16 + r + moff) * HEE + nn;
    hij[o]      = act_tanh(acc0[r] + bi0);
    hij[o + 16] = act_tanh(acc1[r] + bi1);
  }
}

// ---------------- layers 1..3 two-electron GEMM (K=32, residual) -------------
__global__ __launch_bounds__(256) void gemm_hij_kernel(
    float* __restrict__ hij, const bf16_t* __restrict__ Wp,
    const float* __restrict__ bias) {
  const int LD = 40;
  __shared__ float  sAf[128 * 32];   // raw fp32 tile (async target + residual)
  __shared__ bf16_t sAb[128 * 40];
  const int tid = threadIdx.x;
  const int r0 = blockIdx.x * 128;
#if HAVE_ASYNC
  for (int t = tid; t < 1024; t += 256)   // 1024 x 16B = 128 rows x 32 f32
    async_copy16(hij + (size_t)r0 * HEE + t * 4, sAf + t * 4);
  __builtin_amdgcn_s_wait_asynccnt(0);
#else
  for (int idx = tid; idx < 128 * 32; idx += 256)
    sAf[idx] = hij[(size_t)r0 * HEE + idx];
#endif
  __syncthreads();
  for (int idx = tid; idx < 128 * 32; idx += 256) {
    int r = idx >> 5, c = idx & 31;
    sAb[r * LD + c] = (bf16_t)sAf[idx];
  }
  __syncthreads();
  const int wave = tid >> 5, lane = tid & 31;
  v16bf a  = load_a_lds(sAb + (wave * 16) * LD, LD, lane);
  v16bf b0 = load_b_packed(Wp, 0, 0, 2, lane);
  v16bf b1 = load_b_packed(Wp, 0, 1, 2, lane);
  v8f acc0 = {0.f, 0.f, 0.f, 0.f, 0.f, 0.f, 0.f, 0.f};
  v8f acc1 = acc0;
  acc0 = wmma_bf16(a, b0, acc0);
  acc1 = wmma_bf16(a, b1, acc1);
  const int moff = (lane >> 4) << 3, nn = lane & 15;
  float bi0 = bias[nn], bi1 = bias[16 + nn];
#pragma unroll
  for (int r = 0; r < 8; ++r) {
    int lrow = wave * 16 + r + moff;
    size_t o = (size_t)(r0 + lrow) * HEE + nn;
    hij[o]      = act_tanh(acc0[r] + bi0) + sAf[lrow * 32 + nn];
    hij[o + 16] = act_tanh(acc1[r] + bi1) + sAf[lrow * 32 + 16 + nn];
  }
}

// ---------------- host launch ----------------
extern "C" void kernel_launch(void* const* d_in, const int* in_sizes, int n_in,
                              void* d_out, int out_size, void* d_ws, size_t ws_size,
                              hipStream_t stream) {
  const float* pos      = (const float*)d_in[0];
  const float* rnuc     = (const float*)d_in[1];
  const float* We0      = (const float*)d_in[2];   // [56,256]
  const float* be0      = (const float*)d_in[3];   // [256]
  const float* We_rest  = (const float*)d_in[4];   // [3,832,256]
  const float* be_rest  = (const float*)d_in[5];   // [3,256]
  const float* Wee0     = (const float*)d_in[6];   // [4,32]
  const float* bee0     = (const float*)d_in[7];   // [32]
  const float* Wee_rest = (const float*)d_in[8];   // [3,32,32]
  const float* bee_rest = (const float*)d_in[9];   // [3,32]

  const int B = in_sizes[0] / (NEL * 3);           // 4096
  const size_t BN  = (size_t)B * NEL;              // 81920
  const size_t BNN = (size_t)B * NEL * NEL;        // 1,638,400

  float* hi  = (float*)d_out;
  float* hij = hi + BN * HE;

  char* w = (char*)d_ws;
  auto carve = [&](size_t bytes) {
    char* p = w;
    w += (bytes + 255) & ~(size_t)255;
    return p;
  };
  float*  gdown = (float*)carve((size_t)B * HE * 4);
  float*  gup   = (float*)carve((size_t)B * HE * 4);
  float*  gdni  = (float*)carve((size_t)B * NEL * HEE * 4);
  float*  gupi  = (float*)carve((size_t)B * NEL * HEE * 4);
  bf16_t* We0p  = (bf16_t*)carve((size_t)64 * 256 * 2);
  bf16_t* Werp  = (bf16_t*)carve((size_t)3 * FIN * 256 * 2);
  bf16_t* Wee0p = (bf16_t*)carve((size_t)32 * 32 * 2);
  bf16_t* Weerp = (bf16_t*)carve((size_t)3 * 32 * 32 * 2);

  pack_w_kernel<<<(64 * 256 + 255) / 256, 256, 0, stream>>>(We0, We0p, 56, 64, 256);
  for (int l = 0; l < 3; ++l)
    pack_w_kernel<<<(FIN * 256 + 255) / 256, 256, 0, stream>>>(
        We_rest + (size_t)l * FIN * 256, Werp + (size_t)l * FIN * 256, FIN, FIN, 256);
  pack_w_kernel<<<(32 * 32 + 255) / 256, 256, 0, stream>>>(Wee0, Wee0p, 4, 32, 32);
  for (int l = 0; l < 3; ++l)
    pack_w_kernel<<<(32 * 32 + 255) / 256, 256, 0, stream>>>(
        Wee_rest + (size_t)l * 32 * 32, Weerp + (size_t)l * 32 * 32, 32, 32, 32);

  means0_hi_kernel<<<(B * 16 + 255) / 256, 256, 0, stream>>>(pos, rnuc, gdown, gup, B);
  means0_hij_kernel<<<(B * NEL * 4 + 255) / 256, 256, 0, stream>>>(pos, gdni, gupi, B);
  gemm_hi_l0_kernel<<<BN / 16, 256, 0, stream>>>(pos, rnuc, gdown, gup, gdni, gupi,
                                                 We0p, be0, hi);
  gemm_hij_l0_kernel<<<BNN / 128, 256, 0, stream>>>(pos, Wee0p, bee0, hij);

  for (int l = 1; l < NLAY; ++l) {
    means_hi_kernel<<<(B * HE + 255) / 256, 256, 0, stream>>>(hi, gdown, gup, B);
    means_hij_kernel<<<(B * NEL * HEE + 255) / 256, 256, 0, stream>>>(hij, gdni, gupi, B);
    gemm_hi_kernel<<<BN / 16, 256, 0, stream>>>(
        hi, gdown, gup, gdni, gupi, Werp + (size_t)(l - 1) * FIN * 256,
        be_rest + (size_t)(l - 1) * 256);
    gemm_hij_kernel<<<BNN / 128, 256, 0, stream>>>(
        hij, Weerp + (size_t)(l - 1) * 32 * 32, bee_rest + (size_t)(l - 1) * 32);
  }
}
